// TR_GNN_61495341744699
// MI455X (gfx1250) — compile-verified, matrix operand order checked by reading, not run on hardware
//
#include <hip/hip_runtime.h>
#include <hip/hip_bf16.h>
#include <stdint.h>

#define Hdim 64
#define Adim 64
#define NQc 128
#define NRELc 230
#define NENTc 20000
#define Tc 365
#define TDc 16

typedef __attribute__((ext_vector_type(16))) __bf16 v16bf;
typedef __attribute__((ext_vector_type(8)))  float  v8f;

static __device__ __forceinline__ v8f wmma_bf16(v16bf a, v16bf b, v8f c) {
  // D = A(16x32 bf16) x B(32x16 bf16) + C(16x16 f32)
  return __builtin_amdgcn_wmma_f32_16x16x32_bf16(false, a, false, b, (short)0, c, false, false);
}

static __device__ __forceinline__ float sigm(float x) {
  return 1.0f / (1.0f + __expf(-x));
}

// Fragment slice for this lane from one 64-float row (A-role: row M = lane&15;
// B-role: col n = lane&15; identical per-lane k-striping).
// kc selects K window [32kc, 32kc+32). g = lane>>4.
static __device__ __forceinline__ v16bf row_frag(const float* row, int kc, int g) {
  v16bf f;
#pragma unroll
  for (int i = 0; i < 8; ++i) {
    f[i]     = (__bf16)row[32 * kc + 8 * g + i];
    f[8 + i] = (__bf16)row[32 * kc + 16 + 8 * g + i];
  }
  return f;
}

// ---------------- utility kernels ----------------

__global__ void tg_fill_f32(float* p, float v, long long n) {
  long long i = (long long)blockIdx.x * blockDim.x + threadIdx.x;
  long long s = (long long)gridDim.x * blockDim.x;
  for (; i < n; i += s) p[i] = v;
}
__global__ void tg_fill_i32(int* p, int v, long long n) {
  long long i = (long long)blockIdx.x * blockDim.x + threadIdx.x;
  long long s = (long long)gridDim.x * blockDim.x;
  for (; i < n; i += s) p[i] = v;
}
__global__ void tg_scatter_inv(const int* idx, int cnt, int* inv) {
  int i = blockIdx.x * blockDim.x + threadIdx.x;
  if (i < cnt) inv[idx[i]] = i;
}

// tproj[t][a] = sum_d time_emb[t][d] * Wt[d][a]   (K=16, plain)
__global__ void tg_tproj(const float* time_emb, const float* Wt, float* out) {
  int idx = blockIdx.x * blockDim.x + threadIdx.x;
  if (idx >= Tc * Adim) return;
  int t = idx / Adim, a = idx % Adim;
  const float* te = time_emb + (size_t)t * TDc;
  float s = 0.f;
#pragma unroll
  for (int d = 0; d < TDc; ++d) s += te[d] * Wt[d * Adim + a];
  out[idx] = s;
}

// Pre-swizzle a [K x N] (via strides) f32 weight into per-lane-contiguous bf16
// fragments. dst[((t*kchunks+kc)*32 + lane)*16 + i]; value = W[k][16t+(lane&15)],
// k = 32kc + (i<8 ? 8*(lane>>4)+i : 16 + 8*(lane>>4) + (i-8))
__global__ void tg_swz(const float* src, __bf16* dst, int K, int Ntiles, int rs, int cs) {
  int idx = blockIdx.x * blockDim.x + threadIdx.x;
  int kchunks = K / 32;
  int total = Ntiles * kchunks * 32 * 16;
  if (idx >= total) return;
  int i = idx & 15;
  int lane = (idx >> 4) & 31;
  int rest = idx >> 9;
  int kc = rest % kchunks;
  int t = rest / kchunks;
  int g = lane >> 4, nn = lane & 15;
  int k = 32 * kc + ((i < 8) ? (8 * g + i) : (16 + 8 * g + (i - 8)));
  int col = 16 * t + nn;
  dst[idx] = (__bf16)src[(size_t)k * rs + (size_t)col * cs];
}

// out[r][0:64] = src[gidx?gidx[r]:r][0:64] @ W(64x64), WMMA.
__global__ void __launch_bounds__(256) tg_matrow(
    int rows, const float* src, const int* gidx, const __bf16* WF, float* out) {
  const int lane = threadIdx.x & 31;
  const int g = lane >> 4;
  const int nloc = lane & 15;

  v16bf wf[4][2];
#pragma unroll
  for (int t = 0; t < 4; ++t)
#pragma unroll
    for (int kc = 0; kc < 2; ++kc)
      wf[t][kc] = ((const v16bf*)WF)[(t * 2 + kc) * 32 + lane];

  int wave = (blockIdx.x * blockDim.x + threadIdx.x) >> 5;
  int nwaves = (gridDim.x * blockDim.x) >> 5;
  int ntiles = (rows + 15) >> 4;

  for (int tile = wave; tile < ntiles; tile += nwaves) {
    int jbase = tile * 16;
    int ri = jbase + nloc;
    ri = (ri < rows) ? ri : (rows - 1);                 // clamp, branchless
    int si = (gidx != nullptr) ? gidx[ri] : ri;
    const float* row = src + (size_t)si * Hdim;
    v16bf a[2];
#pragma unroll
    for (int kc = 0; kc < 2; ++kc) a[kc] = row_frag(row, kc, g);

    v8f c[4];
#pragma unroll
    for (int t = 0; t < 4; ++t) c[t] = (v8f){0.f, 0.f, 0.f, 0.f, 0.f, 0.f, 0.f, 0.f};
#pragma unroll
    for (int t = 0; t < 4; ++t)
#pragma unroll
      for (int kc = 0; kc < 2; ++kc) c[t] = wmma_bf16(a[kc], wf[t][kc], c[t]);

    // one base pointer per tile; all stores use compile-time immediate offsets
    float* ob = out + (size_t)(jbase + 8 * g) * Hdim + nloc;
    if (jbase + 16 <= rows) {            // full tile: straight-line stores
#pragma unroll
      for (int t = 0; t < 4; ++t)
#pragma unroll
        for (int r = 0; r < 8; ++r) ob[r * Hdim + t * 16] = c[t][r];
    } else {
#pragma unroll
      for (int t = 0; t < 4; ++t)
#pragma unroll
        for (int r = 0; r < 8; ++r)
          if (jbase + 8 * g + r < rows) ob[r * Hdim + t * 16] = c[t][r];
    }
  }
}

// ---------------- edge attention + segment-sum (no matmul; tables precomputed) ----------------
// att[e][:] = relu(HWs[hd] + RWr[rl] + qatt[qid] + tproj[te]); alpha = sigm(att.w_alpha)
// agg[tail] += alpha * (h_prev[hd] + rel_emb[rl])
template <bool USE_HS>
__global__ void __launch_bounds__(256) tg_edge(
    int E,
    const int* qid, const int* head, const int* rel, const int* tail, const int* tim,
    const float* HWs, const float* h_prev, const float* RWr, const float* rel_emb,
    const float* qatt, const float* tproj, const float* wAlpha, float* agg) {
  extern __shared__ float sbuf[];
  float* s_qatt = sbuf;                  // NQc*Adim
  float* s_tproj = sbuf + NQc * Adim;    // Tc*Adim
  for (int i = threadIdx.x; i < NQc * Adim; i += blockDim.x) s_qatt[i] = qatt[i];
  for (int i = threadIdx.x; i < Tc * Adim; i += blockDim.x) s_tproj[i] = tproj[i];
  __syncthreads();

  const int lane = threadIdx.x & 31;
  const float wa0 = wAlpha[lane];
  const float wa1 = wAlpha[lane + 32];

  int wave = (blockIdx.x * blockDim.x + threadIdx.x) >> 5;
  int nwaves = (gridDim.x * blockDim.x) >> 5;
  const long long stride = (long long)nwaves * 32;

  long long base = (long long)wave * 32;
  for (; base + 32 <= E; base += stride) {
    // prefetch next block's index streams (global_prefetch_b8; speculative)
    long long nb = base + stride;
    if (nb < E) {
      __builtin_prefetch(&head[nb + lane], 0, 1);
      __builtin_prefetch(&rel[nb + lane], 0, 1);
      __builtin_prefetch(&qid[nb + lane], 0, 1);
      __builtin_prefetch(&tim[nb + lane], 0, 1);
      __builtin_prefetch(&tail[nb + lane], 0, 1);
    }
    long long e = base + lane;
    int hdv = head[e], rlv = rel[e], qdv = qid[e], tev = tim[e], tlv = tail[e];

#pragma unroll 4
    for (int m = 0; m < 32; ++m) {
      int hd = __shfl(hdv, m, 32);
      int rl = __shfl(rlv, m, 32);
      int qd = __shfl(qdv, m, 32);
      int te = __shfl(tev, m, 32);
      int tl = __shfl(tlv, m, 32);

      float hws0 = 0.f, hws1 = 0.f, hp0 = 0.f, hp1 = 0.f;
      if (USE_HS) {
        const float* hwr = HWs + (size_t)hd * Hdim + lane;
        const float* hpr = h_prev + (size_t)hd * Hdim + lane;
        hws0 = hwr[0]; hws1 = hwr[32];
        hp0 = hpr[0];  hp1 = hpr[32];
      }
      const float* rwr = RWr + (size_t)rl * Hdim + lane;
      const float* rer = rel_emb + (size_t)rl * Hdim + lane;
      const float* sq = s_qatt + qd * Adim + lane;
      const float* st = s_tproj + te * Adim + lane;
      float b0 = hws0 + rwr[0] + sq[0] + st[0];
      float b1 = hws1 + rwr[32] + sq[32] + st[32];
      float part = fmaxf(b0, 0.f) * wa0 + fmaxf(b1, 0.f) * wa1;
      part += __shfl_xor(part, 1, 32);
      part += __shfl_xor(part, 2, 32);
      part += __shfl_xor(part, 4, 32);
      part += __shfl_xor(part, 8, 32);
      part += __shfl_xor(part, 16, 32);
      float alpha = sigm(part);

      float v0 = alpha * (hp0 + rer[0]);
      float v1 = alpha * (hp1 + rer[32]);
      float* arow = agg + (size_t)tl * Hdim + lane;
      atomicAdd(&arow[0], v0);
      atomicAdd(&arow[32], v1);
    }
  }
  // one ragged tail block per grid (at most)
  if (base < E) {
    long long e = base + lane;
    bool ok = e < E;
    int hdv = ok ? head[e] : 0;
    int rlv = ok ? rel[e] : 0;
    int qdv = ok ? qid[e] : 0;
    int tev = ok ? tim[e] : 0;
    int tlv = ok ? tail[e] : 0;
    int cnt = (int)(E - base);
    for (int m = 0; m < cnt; ++m) {
      int hd = __shfl(hdv, m, 32);
      int rl = __shfl(rlv, m, 32);
      int qd = __shfl(qdv, m, 32);
      int te = __shfl(tev, m, 32);
      int tl = __shfl(tlv, m, 32);
      float hws0 = 0.f, hws1 = 0.f, hp0 = 0.f, hp1 = 0.f;
      if (USE_HS) {
        const float* hwr = HWs + (size_t)hd * Hdim + lane;
        const float* hpr = h_prev + (size_t)hd * Hdim + lane;
        hws0 = hwr[0]; hws1 = hwr[32];
        hp0 = hpr[0];  hp1 = hpr[32];
      }
      const float* rwr = RWr + (size_t)rl * Hdim + lane;
      const float* rer = rel_emb + (size_t)rl * Hdim + lane;
      const float* sq = s_qatt + qd * Adim + lane;
      const float* st = s_tproj + te * Adim + lane;
      float b0 = hws0 + rwr[0] + sq[0] + st[0];
      float b1 = hws1 + rwr[32] + sq[32] + st[32];
      float part = fmaxf(b0, 0.f) * wa0 + fmaxf(b1, 0.f) * wa1;
      part += __shfl_xor(part, 1, 32);
      part += __shfl_xor(part, 2, 32);
      part += __shfl_xor(part, 4, 32);
      part += __shfl_xor(part, 8, 32);
      part += __shfl_xor(part, 16, 32);
      float alpha = sigm(part);
      float v0 = alpha * (hp0 + rer[0]);
      float v1 = alpha * (hp1 + rer[32]);
      float* arow = agg + (size_t)tl * Hdim + lane;
      atomicAdd(&arow[0], v0);
      atomicAdd(&arow[32], v1);
    }
  }
}

// ---------------- node GRU kernel (all-WMMA, no LDS transpose) ----------------
// xT = Wh^T @ agg^T gives x directly in A-fragment layout for the gate matmuls.
__global__ void __launch_bounds__(256) tg_node(
    int n, const float* agg, const int* inv, const float* h_prev, const float* zrow,
    const __bf16* WhF, const __bf16* WihF, const __bf16* WhhF,
    const float* bih, const float* bhh, float* hid_out) {
  const int lane = threadIdx.x & 31;
  const int g = lane >> 4;
  const int nloc = lane & 15;

  // hoist per-lane bias terms (col = 16t + nloc)
  float bsum0[4], bsum1[4], bi2[4], bh2[4];
#pragma unroll
  for (int t = 0; t < 4; ++t) {
    int col = t * 16 + nloc;
    bsum0[t] = bih[col] + bhh[col];
    bsum1[t] = bih[64 + col] + bhh[64 + col];
    bi2[t] = bih[128 + col];
    bh2[t] = bhh[128 + col];
  }

  int wave = (blockIdx.x * blockDim.x + threadIdx.x) >> 5;
  int nwaves = (gridDim.x * blockDim.x) >> 5;
  int ntiles = (n + 15) >> 4;

  for (int tile = wave; tile < ntiles; tile += nwaves) {
    int jbase = tile * 16;
    int jcl = jbase + lane;
    jcl = (jcl < n) ? jcl : (n - 1);  // clamp
    int iv = (lane < 16) ? inv[jcl] : -1;

    int ncl = jbase + nloc;
    ncl = (ncl < n) ? ncl : (n - 1);  // clamp

    // B-frags of agg^T (per-lane: node nloc's row, k-striped)
    const float* arow = agg + (size_t)ncl * Hdim;
    v16bf aggB[2];
#pragma unroll
    for (int kc = 0; kc < 2; ++kc) aggB[kc] = row_frag(arow, kc, g);

    // xT[tt] tiles: A = Wh^T row-tile tt (same swizzled buffer), B = agg^T
    v8f xt[4];
#pragma unroll
    for (int t = 0; t < 4; ++t) xt[t] = (v8f){0.f, 0.f, 0.f, 0.f, 0.f, 0.f, 0.f, 0.f};
#pragma unroll
    for (int t = 0; t < 4; ++t)
#pragma unroll
      for (int kc = 0; kc < 2; ++kc)
        xt[t] = wmma_bf16(((const v16bf*)WhF)[(t * 2 + kc) * 32 + lane], aggB[kc], xt[t]);

    // xT D-layout == x A-fragment layout
    v16bf xA[2];
#pragma unroll
    for (int kc = 0; kc < 2; ++kc) {
      v16bf f;
#pragma unroll
      for (int i = 0; i < 8; ++i) {
        f[i]     = (__bf16)fmaxf(xt[2 * kc][i], 0.f);
        f[8 + i] = (__bf16)fmaxf(xt[2 * kc + 1][i], 0.f);
      }
      xA[kc] = f;
    }

    // h0n A-frags: gather via inverse index; zero row for absent nodes
    int iv_m = __shfl(iv, nloc, 32);
    const float* h0row = (iv_m >= 0) ? (h_prev + (size_t)iv_m * Hdim) : zrow;
    v16bf h0A[2];
#pragma unroll
    for (int kc = 0; kc < 2; ++kc) h0A[kc] = row_frag(h0row, kc, g);

    // r/z gates: gi+gh fused into shared accumulators (8 tiles)
    v8f rz[8];
#pragma unroll
    for (int t = 0; t < 8; ++t) rz[t] = (v8f){0.f, 0.f, 0.f, 0.f, 0.f, 0.f, 0.f, 0.f};
#pragma unroll
    for (int t = 0; t < 8; ++t)
#pragma unroll
      for (int kc = 0; kc < 2; ++kc) {
        rz[t] = wmma_bf16(xA[kc], ((const v16bf*)WihF)[(t * 2 + kc) * 32 + lane], rz[t]);
        rz[t] = wmma_bf16(h0A[kc], ((const v16bf*)WhhF)[(t * 2 + kc) * 32 + lane], rz[t]);
      }

    float rg[4][8], zg[4][8];
#pragma unroll
    for (int t = 0; t < 4; ++t)
#pragma unroll
      for (int r = 0; r < 8; ++r) {
        rg[t][r] = sigm(rz[t][r] + bsum0[t]);
        zg[t][r] = sigm(rz[t + 4][r] + bsum1[t]);
      }

    // n gate: need gi_n and gh_n separately
    v8f gin[4], ghn[4];
#pragma unroll
    for (int t = 0; t < 4; ++t) {
      gin[t] = (v8f){0.f, 0.f, 0.f, 0.f, 0.f, 0.f, 0.f, 0.f};
      ghn[t] = (v8f){0.f, 0.f, 0.f, 0.f, 0.f, 0.f, 0.f, 0.f};
    }
#pragma unroll
    for (int t = 0; t < 4; ++t)
#pragma unroll
      for (int kc = 0; kc < 2; ++kc) {
        gin[t] = wmma_bf16(xA[kc], ((const v16bf*)WihF)[((8 + t) * 2 + kc) * 32 + lane], gin[t]);
        ghn[t] = wmma_bf16(h0A[kc], ((const v16bf*)WhhF)[((8 + t) * 2 + kc) * 32 + lane], ghn[t]);
      }

    // h0n rows for the final combine (D layout), zero-row for absent nodes
    const float* h0r[8];
#pragma unroll
    for (int r = 0; r < 8; ++r) {
      int ivr = __shfl(iv, r + 8 * g, 32);
      h0r[r] = ((ivr >= 0) ? (h_prev + (size_t)ivr * Hdim) : zrow) + nloc;
    }

    // one base pointer per tile; stores use immediate offsets
    float* ob = hid_out + (size_t)(jbase + 8 * g) * Hdim + nloc;
    bool full = (jbase + 16) <= n;
    if (full) {
#pragma unroll
      for (int t = 0; t < 4; ++t)
#pragma unroll
        for (int r = 0; r < 8; ++r) {
          float h0v = h0r[r][t * 16];
          float ng = tanhf(gin[t][r] + bi2[t] + rg[t][r] * (ghn[t][r] + bh2[t]));
          ob[r * Hdim + t * 16] = (1.f - zg[t][r]) * ng + zg[t][r] * h0v;
        }
    } else {
#pragma unroll
      for (int t = 0; t < 4; ++t)
#pragma unroll
        for (int r = 0; r < 8; ++r) {
          if (jbase + 8 * g + r < n) {
            float h0v = h0r[r][t * 16];
            float ng = tanhf(gin[t][r] + bi2[t] + rg[t][r] * (ghn[t][r] + bh2[t]));
            ob[r * Hdim + t * 16] = (1.f - zg[t][r]) * ng + zg[t][r] * h0v;
          }
        }
    }
  }
}

// ---------------- final scoring ----------------
__global__ void tg_score(int n2, const float* hid, const float* Wf,
                         const int* nq, const int* ne, float* out) {
  int i = blockIdx.x * blockDim.x + threadIdx.x;
  if (i >= n2) return;
  const float* row = hid + (size_t)i * Hdim;
  float s = 0.f;
#pragma unroll
  for (int h = 0; h < Hdim; ++h) s += row[h] * Wf[h];
  out[(size_t)nq[i] * NENTc + ne[i]] = s;
}

// ---------------- host launcher ----------------
extern "C" void kernel_launch(void* const* d_in, const int* in_sizes, int n_in,
                              void* d_out, int out_size, void* d_ws, size_t ws_size,
                              hipStream_t stream) {
  (void)n_in; (void)ws_size;
  const int* relation = (const int*)d_in[1];
  const int* qidL[3]  = {(const int*)d_in[2],  (const int*)d_in[8],  (const int*)d_in[14]};
  const int* headL[3] = {(const int*)d_in[3],  (const int*)d_in[9],  (const int*)d_in[15]};
  const int* relL[3]  = {(const int*)d_in[4],  (const int*)d_in[10], (const int*)d_in[16]};
  const int* tailL[3] = {(const int*)d_in[5],  (const int*)d_in[11], (const int*)d_in[17]};
  const int* timL[3]  = {(const int*)d_in[6],  (const int*)d_in[12], (const int*)d_in[18]};
  const int* idxL[3]  = {(const int*)d_in[7],  (const int*)d_in[13], (const int*)d_in[19]};
  const int* node_q = (const int*)d_in[20];
  const int* node_e = (const int*)d_in[21];
  const float* rel_embs = (const float*)d_in[25];
  const float* Ws = (const float*)d_in[26];
  const float* Wr = (const float*)d_in[27];
  const float* Wq = (const float*)d_in[28];
  const float* Wt = (const float*)d_in[29];
  const float* w_alpha = (const float*)d_in[30];
  const float* Wh = (const float*)d_in[31];
  const float* time_emb = (const float*)d_in[32];
  const float* Wih = (const float*)d_in[33];
  const float* Whh = (const float*)d_in[34];
  const float* bih = (const float*)d_in[35];
  const float* bhh = (const float*)d_in[36];
  const float* Wfinal = (const float*)d_in[37];

  int E[3]     = {in_sizes[2], in_sizes[8], in_sizes[14]};
  int prevN[3] = {in_sizes[7], in_sizes[13], in_sizes[19]};   // {NQ, N0, N1}
  int NL[3]    = {in_sizes[13], in_sizes[19], in_sizes[20]};  // {N0, N1, N2}
  int n2 = in_sizes[20];

  // carve workspace
  char* p = (char*)d_ws;
  auto carve = [&](size_t bytes) -> void* {
    void* r = (void*)p;
    p += (bytes + 255) & ~(size_t)255;
    return r;
  };
  float* hidA = (float*)carve((size_t)NL[2] * Hdim * 4);
  float* hidB = (float*)carve((size_t)NL[1] * Hdim * 4);
  float* agg  = (float*)carve((size_t)NL[2] * Hdim * 4);
  float* HWs  = (float*)carve((size_t)NL[1] * Hdim * 4);  // max prev rows = N1
  float* zeroHid = (float*)carve((size_t)NQc * Hdim * 4);
  int* inv = (int*)carve((size_t)NL[2] * 4);
  float* qatt = (float*)carve((size_t)3 * NQc * Adim * 4);
  float* tproj = (float*)carve((size_t)3 * Tc * Adim * 4);
  float* RWr = (float*)carve((size_t)NRELc * Hdim * 4);
  __bf16* WsF = (__bf16*)carve((size_t)3 * 4096 * 2);
  __bf16* WrF = (__bf16*)carve((size_t)3 * 4096 * 2);
  __bf16* WqF = (__bf16*)carve((size_t)3 * 4096 * 2);
  __bf16* WhF = (__bf16*)carve((size_t)3 * 4096 * 2);
  __bf16* WihF = (__bf16*)carve((size_t)12288 * 2);
  __bf16* WhhF = (__bf16*)carve((size_t)12288 * 2);

  float* outf = (float*)d_out;
  const size_t EDGE_LDS = (size_t)(NQc * Adim + Tc * Adim) * sizeof(float);  // ~123 KB
  hipFuncSetAttribute(reinterpret_cast<const void*>(tg_edge<true>),
                      hipFuncAttributeMaxDynamicSharedMemorySize, (int)EDGE_LDS);
  hipFuncSetAttribute(reinterpret_cast<const void*>(tg_edge<false>),
                      hipFuncAttributeMaxDynamicSharedMemorySize, (int)EDGE_LDS);

  tg_fill_f32<<<2048, 256, 0, stream>>>(outf, 0.f, (long long)out_size);
  tg_fill_f32<<<32, 256, 0, stream>>>(zeroHid, 0.f, (long long)NQc * Hdim);

  for (int l = 0; l < 3; ++l) {
    tg_tproj<<<(Tc * Adim + 255) / 256, 256, 0, stream>>>(
        time_emb, Wt + (size_t)l * TDc * Adim, tproj + (size_t)l * Tc * Adim);
    tg_swz<<<16, 256, 0, stream>>>(Ws + (size_t)l * Hdim * Adim, WsF + (size_t)l * 4096, 64, 4, 64, 1);
    tg_swz<<<16, 256, 0, stream>>>(Wr + (size_t)l * Hdim * Adim, WrF + (size_t)l * 4096, 64, 4, 64, 1);
    tg_swz<<<16, 256, 0, stream>>>(Wq + (size_t)l * Hdim * Adim, WqF + (size_t)l * 4096, 64, 4, 64, 1);
    tg_swz<<<16, 256, 0, stream>>>(Wh + (size_t)l * Hdim * Hdim, WhF + (size_t)l * 4096, 64, 4, 64, 1);
  }
  // gi = x @ Wih^T : B[k][n] = Wih[n][k]  (rs=1, cs=64)
  tg_swz<<<48, 256, 0, stream>>>(Wih, WihF, 64, 12, 1, 64);
  tg_swz<<<48, 256, 0, stream>>>(Whh, WhhF, 64, 12, 1, 64);

  // q_att[l] = (rel_emb_l[relation]) @ Wq[l]   (WMMA, gathered rows)
  for (int l = 0; l < 3; ++l)
    tg_matrow<<<1, 256, 0, stream>>>(NQc, rel_embs + (size_t)l * NRELc * Hdim, relation,
                                     WqF + (size_t)l * 4096, qatt + (size_t)l * NQc * Adim);

  const float* hp[3] = {zeroHid, hidA, hidB};
  float* ho[3] = {hidA, hidB, hidA};

  for (int l = 0; l < 3; ++l) {
    // RWr = rel_emb @ Wr  (230 rows, WMMA)
    tg_matrow<<<2, 256, 0, stream>>>(NRELc, rel_embs + (size_t)l * NRELc * Hdim, nullptr,
                                     WrF + (size_t)l * 4096, RWr);
    // HWs = h_prev @ Ws  (node-level hoist of the per-edge matmul)
    if (l > 0)
      tg_matrow<<<512, 256, 0, stream>>>(prevN[l], hp[l], nullptr,
                                         WsF + (size_t)l * 4096, HWs);
    tg_fill_f32<<<2048, 256, 0, stream>>>(agg, 0.f, (long long)NL[l] * Hdim);
    if (l > 0) {
      tg_edge<true><<<768, 256, EDGE_LDS, stream>>>(
          E[l], qidL[l], headL[l], relL[l], tailL[l], timL[l],
          HWs, hp[l], RWr, rel_embs + (size_t)l * NRELc * Hdim,
          qatt + (size_t)l * NQc * Adim, tproj + (size_t)l * Tc * Adim,
          w_alpha + (size_t)l * Adim, agg);
    } else {
      tg_edge<false><<<768, 256, EDGE_LDS, stream>>>(
          E[l], qidL[l], headL[l], relL[l], tailL[l], timL[l],
          HWs, hp[l], RWr, rel_embs + (size_t)l * NRELc * Hdim,
          qatt + (size_t)l * NQc * Adim, tproj + (size_t)l * Tc * Adim,
          w_alpha + (size_t)l * Adim, agg);
    }
    tg_fill_i32<<<2048, 256, 0, stream>>>(inv, -1, (long long)NL[l]);
    tg_scatter_inv<<<(prevN[l] + 255) / 256, 256, 0, stream>>>(idxL[l], prevN[l], inv);
    tg_node<<<512, 256, 0, stream>>>(
        NL[l], agg, inv, hp[l], zeroHid, WhF + (size_t)l * 4096, WihF, WhhF, bih, bhh, ho[l]);
  }

  tg_score<<<(n2 + 255) / 256, 256, 0, stream>>>(n2, hidA, Wfinal, node_q, node_e, outf);
}
